// CapsuleLayer_42176578847005
// MI455X (gfx1250) — matrix operand order repaired and checked by reading, not compiled, not run on hardware
//
#include <hip/hip_runtime.h>
#include <math.h>

// ---------------------------------------------------------------------------
// CapsuleLayer dynamic routing for MI455X (gfx1250, wave32, WMMA).
//   x [64,512,128] f32, W [32,512,64,128] f32  ->  out [64,32] f32
// Phase 1: u[b,n,i,o] via v_wmma_f32_16x16x32_bf16 (bf16 inputs, f32 acc),
//          stored bf16 (128 MB -> L2-resident for the routing passes).
// Phase 2: 3 routing iterations (softmax over n / matvec / agreement) + norm.
// Workspace requirement: 128MB (u) + 4MB (b) + 4MB (c) + 512KB (outputs).
// ---------------------------------------------------------------------------

typedef __attribute__((ext_vector_type(4)))  float  f32x4;
typedef __attribute__((ext_vector_type(8)))  float  v8f;
typedef __attribute__((ext_vector_type(16))) __bf16 v16bf;
typedef __attribute__((ext_vector_type(8)))  __bf16 v8bf;
typedef __attribute__((ext_vector_type(4)))  __bf16 v4bf;

#define B_  64
#define N_  32
#define I_  512
#define O_  64
#define D_  128

__device__ __forceinline__ v16bf pack_bf16x16(f32x4 a, f32x4 b, f32x4 c, f32x4 d) {
    v16bf r;
    r[0]=(__bf16)a[0];  r[1]=(__bf16)a[1];  r[2]=(__bf16)a[2];  r[3]=(__bf16)a[3];
    r[4]=(__bf16)b[0];  r[5]=(__bf16)b[1];  r[6]=(__bf16)b[2];  r[7]=(__bf16)b[3];
    r[8]=(__bf16)c[0];  r[9]=(__bf16)c[1];  r[10]=(__bf16)c[2]; r[11]=(__bf16)c[3];
    r[12]=(__bf16)d[0]; r[13]=(__bf16)d[1]; r[14]=(__bf16)d[2]; r[15]=(__bf16)d[3];
    return r;
}

// ---------------------------------------------------------------------------
// Phase 1: u[b,n,i,o] = sum_d x[b,i,d] * W[n,i,o,d]
// One wave per (n, i, o_tile); wave sweeps all 4 batch tiles so every W
// element is loaded from HBM exactly once (non-temporal). 65536 waves total.
// ---------------------------------------------------------------------------
__global__ __launch_bounds__(256) void caps_gemm_u(
    const float* __restrict__ x, const float* __restrict__ W,
    __bf16* __restrict__ u)
{
    const int lane = threadIdx.x & 31;
    const int wid  = blockIdx.x * 8 + (threadIdx.x >> 5);  // 0..65535
    const int o_t  = wid & 3;
    const int i_i  = (wid >> 2) & (I_ - 1);
    const int n_i  = wid >> 11;
    const int col  = lane & 15;
    const int hi   = lane >> 4;          // 0: lanes 0-15, 1: lanes 16-31

    // A-fragment source: W row (o = o_t*16 + col), d contiguous
    const float* Wrow = W + ((size_t)(n_i * I_ + i_i) * O_ + (o_t * 16 + col)) * D_;
    // B-fragment source: x[b, i_i, d]
    const float* xb = x + (size_t)i_i * D_;

    v8f acc[4] = {v8f{}, v8f{}, v8f{}, v8f{}};

    #pragma unroll
    for (int k = 0; k < 4; ++k) {        // K-steps of 32 over D=128
        const int kb = k * 32;
        // A layout: halves 0-7 -> K = kb + ka+0..7 ; halves 8-15 -> K = kb+16+ka..+7
        const int ka = kb + hi * 8;
        f32x4 a0 = __builtin_nontemporal_load((const f32x4*)(Wrow + ka));
        f32x4 a1 = __builtin_nontemporal_load((const f32x4*)(Wrow + ka + 4));
        f32x4 a2 = __builtin_nontemporal_load((const f32x4*)(Wrow + ka + 16));
        f32x4 a3 = __builtin_nontemporal_load((const f32x4*)(Wrow + ka + 20));
        const v16bf A = pack_bf16x16(a0, a1, a2, a3);

        // B layout: same as A: halves 0-7 -> K = kx+0..7 ; halves 8-15 -> K = kx+16..23
        const int kx = kb + hi * 8;
        #pragma unroll
        for (int bt = 0; bt < 4; ++bt) {
            const float* xr = xb + (size_t)(bt * 16 + col) * (I_ * D_) + kx;
            f32x4 b0 = *(const f32x4*)(xr);
            f32x4 b1 = *(const f32x4*)(xr + 4);
            f32x4 b2 = *(const f32x4*)(xr + 16);
            f32x4 b3 = *(const f32x4*)(xr + 20);
            const v16bf Bf = pack_bf16x16(b0, b1, b2, b3);
            acc[bt] = __builtin_amdgcn_wmma_f32_16x16x32_bf16(
                false, A, false, Bf, (short)0, acc[bt], false, false);
        }
    }

    // C/D layout: lane col = batch column, VGPR r -> o row = o_t*16 + hi*8 + r
    const int o0 = o_t * 16 + hi * 8;
    #pragma unroll
    for (int bt = 0; bt < 4; ++bt) {
        const int b = bt * 16 + col;
        v8bf ov;
        #pragma unroll
        for (int r = 0; r < 8; ++r) ov[r] = (__bf16)acc[bt][r];
        *(v8bf*)(u + ((size_t)(b * N_ + n_i) * I_ + i_i) * O_ + o0) = ov;
    }
}

// ---------------------------------------------------------------------------
// outputs[b,n,o] = sum_i c[b,n,i] * u[b,n,i,o]   (c uniform 1/N on iter 0)
// One block of 64 threads per (b,n); lane o streams the 512x64 bf16 slice.
// ---------------------------------------------------------------------------
__global__ __launch_bounds__(64) void caps_outputs(
    const __bf16* __restrict__ u, const float* __restrict__ c,
    float* __restrict__ outs, int uniform)
{
    const int bn = blockIdx.x;           // b*N_ + n
    const int o  = threadIdx.x;
    const __bf16* up = u + (size_t)bn * I_ * O_ + o;
    float acc = 0.f;
    if (uniform) {
        #pragma unroll 8
        for (int i = 0; i < I_; ++i) acc += (float)up[(size_t)i * O_];
        acc *= (1.0f / (float)N_);
    } else {
        const float* cp = c + (size_t)bn * I_;
        #pragma unroll 8
        for (int i = 0; i < I_; ++i) acc += cp[i] * (float)up[(size_t)i * O_];
    }
    outs[(size_t)bn * O_ + o] = acc;
}

// ---------------------------------------------------------------------------
// bcoef[b,n,i] = sum_o outputs[b,n,o] * u[b,n,i,o]
// ---------------------------------------------------------------------------
__global__ __launch_bounds__(256) void caps_bupdate(
    const __bf16* __restrict__ u, const float* __restrict__ outs,
    float* __restrict__ bco)
{
    const int bn = blockIdx.x;
    __shared__ float so[O_];
    if (threadIdx.x < O_) so[threadIdx.x] = outs[(size_t)bn * O_ + threadIdx.x];
    __syncthreads();
    for (int i = threadIdx.x; i < I_; i += 256) {
        const __bf16* up = u + ((size_t)bn * I_ + i) * O_;
        float acc = 0.f;
        #pragma unroll
        for (int o4 = 0; o4 < O_; o4 += 4) {
            v4bf uv = *(const v4bf*)(up + o4);
            acc += so[o4]     * (float)uv[0] + so[o4 + 1] * (float)uv[1]
                 + so[o4 + 2] * (float)uv[2] + so[o4 + 3] * (float)uv[3];
        }
        bco[(size_t)bn * I_ + i] = acc;
    }
}

// ---------------------------------------------------------------------------
// c[b,n,i] = softmax over n of bcoef[b,n,i]   (one thread per (b,i))
// ---------------------------------------------------------------------------
__global__ __launch_bounds__(256) void caps_softmax(
    const float* __restrict__ bco, float* __restrict__ c)
{
    const int t = blockIdx.x * 256 + threadIdx.x;   // 0..B_*I_-1
    if (t >= B_ * I_) return;
    const int b = t >> 9;
    const int i = t & (I_ - 1);
    const float* p = bco + (size_t)b * N_ * I_ + i;
    float v[N_];
    float m = -3.0e38f;
    #pragma unroll
    for (int n = 0; n < N_; ++n) { v[n] = p[(size_t)n * I_]; m = fmaxf(m, v[n]); }
    float s = 0.f;
    #pragma unroll
    for (int n = 0; n < N_; ++n) { v[n] = __expf(v[n] - m); s += v[n]; }
    const float inv = 1.0f / s;
    float* q = c + (size_t)b * N_ * I_ + i;
    #pragma unroll
    for (int n = 0; n < N_; ++n) q[(size_t)n * I_] = v[n] * inv;
}

// ---------------------------------------------------------------------------
// out[b,n] = || outputs[b,n,:] ||
// ---------------------------------------------------------------------------
__global__ __launch_bounds__(256) void caps_norm(
    const float* __restrict__ outs, float* __restrict__ out)
{
    const int t = blockIdx.x * 256 + threadIdx.x;
    if (t >= B_ * N_) return;
    const float* p = outs + (size_t)t * O_;
    float s = 0.f;
    #pragma unroll
    for (int o = 0; o < O_; ++o) s += p[o] * p[o];
    out[t] = sqrtf(s);
}

// ---------------------------------------------------------------------------
extern "C" void kernel_launch(void* const* d_in, const int* in_sizes, int n_in,
                              void* d_out, int out_size, void* d_ws, size_t ws_size,
                              hipStream_t stream)
{
    (void)in_sizes; (void)n_in; (void)out_size; (void)ws_size;
    const float* x = (const float*)d_in[0];   // [64,512,128]
    const float* W = (const float*)d_in[1];   // [32,512,64,128]
    float* out = (float*)d_out;               // [64,32]

    char* ws = (char*)d_ws;
    __bf16* u  = (__bf16*)ws;                                   // 128 MiB
    size_t off = (size_t)B_ * N_ * I_ * O_ * sizeof(__bf16);
    float* bco = (float*)(ws + off);  off += (size_t)B_ * N_ * I_ * sizeof(float);
    float* c   = (float*)(ws + off);  off += (size_t)B_ * N_ * I_ * sizeof(float);
    float* outs = (float*)(ws + off);                           // 512 KiB

    // Phase 1: u via WMMA (65536 waves = 8192 blocks x 8 waves)
    caps_gemm_u<<<8192, 256, 0, stream>>>(x, W, u);

    // Routing iteration 0 (softmax(0) == uniform 1/N)
    caps_outputs<<<B_ * N_, 64, 0, stream>>>(u, nullptr, outs, 1);
    caps_bupdate<<<B_ * N_, 256, 0, stream>>>(u, outs, bco);
    // Routing iteration 1
    caps_softmax<<<(B_ * I_) / 256, 256, 0, stream>>>(bco, c);
    caps_outputs<<<B_ * N_, 64, 0, stream>>>(u, c, outs, 0);
    caps_bupdate<<<B_ * N_, 256, 0, stream>>>(u, outs, bco);
    // Routing iteration 2
    caps_softmax<<<(B_ * I_) / 256, 256, 0, stream>>>(bco, c);
    caps_outputs<<<B_ * N_, 64, 0, stream>>>(u, c, outs, 0);

    // Capsule lengths
    caps_norm<<<(B_ * N_ + 255) / 256, 256, 0, stream>>>(outs, out);
}
